// SimpleTransformerLayer_601295422163
// MI455X (gfx1250) — matrix operand order repaired and checked by reading, not compiled, not run on hardware
//
#include <hip/hip_runtime.h>
#include <math.h>

// ---------------------------------------------------------------------------
// Types for CDNA5 WMMA (gfx1250, wave32)
// ---------------------------------------------------------------------------
typedef __attribute__((ext_vector_type(16))) __bf16 v16bf;
typedef __attribute__((ext_vector_type(8)))  float  v8f;
typedef __attribute__((ext_vector_type(4)))  int    v4i;

union BFrag {            // 16 bf16 elements = 8 dwords (one WMMA A or B operand)
    unsigned u[8];
    v16bf    v;
};

__device__ inline v8f wmma_bf16(const BFrag& a, const BFrag& b, v8f c) {
    return __builtin_amdgcn_wmma_f32_16x16x32_bf16(
        /*neg_a=*/false, a.v, /*neg_b=*/false, b.v,
        /*c_mod=*/(short)0, c, /*reuse_a=*/false, /*reuse_b=*/false);
}

__device__ inline unsigned short f2bf(float f) {
    unsigned u = __builtin_bit_cast(unsigned, f);
    u += 0x7fffu + ((u >> 16) & 1u);        // round-to-nearest-even
    return (unsigned short)(u >> 16);
}

__device__ inline float halfmax(float v) {   // reduce over 16-lane half (wave32)
    for (int m = 1; m < 16; m <<= 1) v = fmaxf(v, __shfl_xor(v, m, 32));
    return v;
}
__device__ inline float halfsum(float v) {
    for (int m = 1; m < 16; m <<= 1) v += __shfl_xor(v, m, 32);
    return v;
}

#define HID 1024
#define SEQ 2048
#define NB  2

// ---------------------------------------------------------------------------
// Async global -> LDS copy (CDNA5 GLOBAL_LOAD_ASYNC_TO_LDS_B128, ASYNCcnt).
// Builtin signature (from clang diagnostic): (v4i AS1*, v4i AS3*, Ii, Ii).
// Guarded: falls back to a synchronous copy if the builtin is unavailable.
// ---------------------------------------------------------------------------
#if __has_builtin(__builtin_amdgcn_global_load_async_to_lds_b128) && \
    __has_builtin(__builtin_amdgcn_s_wait_asynccnt)
#define USE_ASYNC 1
typedef __attribute__((address_space(1))) v4i g_v4i;
typedef __attribute__((address_space(3))) v4i l_v4i;
__device__ inline void cp_async16(const void* g, void* l) {
    // generic->AS1: global generic addr == global addr; generic LDS addr
    // truncates to the 32-bit LDS offset for AS3.
    __builtin_amdgcn_global_load_async_to_lds_b128(
        (g_v4i*)(size_t)g, (l_v4i*)(unsigned)(size_t)l, 0, 0);
}
#define ASYNC_WAIT(n) __builtin_amdgcn_s_wait_asynccnt(n)
#else
__device__ inline void cp_async16(const void* g, void* l) {
    *(uint4*)l = *(const uint4*)g;
}
#define ASYNC_WAIT(n)
#endif

// ---------------------------------------------------------------------------
// fp32 -> bf16 weight conversion
// ---------------------------------------------------------------------------
__global__ void cvt_bf16_kernel(const float* __restrict__ src,
                                unsigned short* __restrict__ dst, int n) {
    int i = blockIdx.x * 256 + threadIdx.x;
    if (i < n) dst[i] = f2bf(src[i]);
}

// ---------------------------------------------------------------------------
// Row LayerNorm: 1024 cols, 256 threads/row, bf16 output
// ---------------------------------------------------------------------------
__global__ void __launch_bounds__(256)
ln_kernel(const float* __restrict__ x, const float* __restrict__ g,
          const float* __restrict__ be, unsigned short* __restrict__ out) {
    __shared__ float red[256];
    int row = blockIdx.x;
    const float* xr = x + (size_t)row * HID;
    float loc[4];
    float s = 0.f;
    for (int i = 0; i < 4; ++i) { loc[i] = xr[threadIdx.x + 256 * i]; s += loc[i]; }
    red[threadIdx.x] = s; __syncthreads();
    for (int off = 128; off > 0; off >>= 1) {
        if (threadIdx.x < off) red[threadIdx.x] += red[threadIdx.x + off];
        __syncthreads();
    }
    float mu = red[0] * (1.f / HID);
    __syncthreads();
    float vs = 0.f;
    for (int i = 0; i < 4; ++i) { float d = loc[i] - mu; vs += d * d; }
    red[threadIdx.x] = vs; __syncthreads();
    for (int off = 128; off > 0; off >>= 1) {
        if (threadIdx.x < off) red[threadIdx.x] += red[threadIdx.x + off];
        __syncthreads();
    }
    float rsig = rsqrtf(red[0] * (1.f / HID) + 1e-5f);
    for (int i = 0; i < 4; ++i) {
        int c = threadIdx.x + 256 * i;
        out[(size_t)row * HID + c] = f2bf((loc[i] - mu) * rsig * g[c] + be[c]);
    }
}

// ---------------------------------------------------------------------------
// Generic WMMA GEMM:  C[M,N] = A[MxK](bf16) * W[NxK]^T(bf16) + bias
//   mode 0: outBf = bf16(C)
//   mode 1: outF  = C + res           (fp32)
//   mode 2: outBf = bf16(gelu(C))     (exact erf GELU)
// WG tile 128x64, 8 waves in 4(M)x2(N); wave tile 32x32 -> 4 WMMAs per K step.
// Double-buffered LDS tiles filled by async global->LDS copies.
// ---------------------------------------------------------------------------
__global__ void __launch_bounds__(256)
gemm_kernel(const unsigned short* __restrict__ A,
            const unsigned short* __restrict__ W,
            const float* __restrict__ bias,
            unsigned short* __restrict__ outBf,
            float* __restrict__ outF,
            const float* __restrict__ res,
            int M, int N, int K, int mode) {
    __shared__ unsigned short At[2][128][40];
    __shared__ unsigned short Wt[2][64][40];
    int n0 = blockIdx.x * 64, m0 = blockIdx.y * 128;
    int tid = threadIdx.x;
    int w = tid >> 5, lane = tid & 31, l16 = lane & 15, hi = lane >> 4;
    int wm = w & 3, wn = w >> 2;                 // wave grid 4(M) x 2(N)
    int ldr = tid >> 2, ldc = (tid & 3) * 8;     // cooperative tile loads

    int steps = K >> 5;
    v8f acc[2][2] = {};

    auto issue = [&](int step, int buf) {
        int k0 = step * 32;
        cp_async16(A + (size_t)(m0 + ldr) * K + k0 + ldc,      &At[buf][ldr][ldc]);
        cp_async16(A + (size_t)(m0 + ldr + 64) * K + k0 + ldc, &At[buf][ldr + 64][ldc]);
        cp_async16(W + (size_t)(n0 + ldr) * K + k0 + ldc,      &Wt[buf][ldr][ldc]);
    };

    issue(0, 0);
    for (int i = 0; i < steps; ++i) {
        int buf = i & 1;
        __syncthreads();                         // buf^1 reads (iter i-1) done
        if (i + 1 < steps) { issue(i + 1, buf ^ 1); ASYNC_WAIT(3); }
        else               { ASYNC_WAIT(0); }
        __syncthreads();                         // step-i tile visible to all waves

        BFrag af[2], wf[2];
        for (int mt = 0; mt < 2; ++mt)           // A frag per ISA 16-bit A layout
            for (int r = 0; r < 8; ++r) {
                int kk = (r < 4 ? 0 : 16) + (hi ? 8 : 0) + 2 * (r & 3);
                af[mt].u[r] = *(const unsigned*)&At[buf][wm * 32 + mt * 16 + l16][kk];
            }
        int kb = hi ? 8 : 0;                     // B frag: same K order as A frag
        for (int nt = 0; nt < 2; ++nt)
            for (int r = 0; r < 8; ++r)
                wf[nt].u[r] = *(const unsigned*)&Wt[buf][wn * 32 + nt * 16 + l16][(r < 4 ? 0 : 16) + kb + 2 * (r & 3)];
        for (int mt = 0; mt < 2; ++mt)
            for (int nt = 0; nt < 2; ++nt)
                acc[mt][nt] = wmma_bf16(af[mt], wf[nt], acc[mt][nt]);
    }

    for (int mt = 0; mt < 2; ++mt)
        for (int nt = 0; nt < 2; ++nt) {
            int n = n0 + wn * 32 + nt * 16 + l16;
            float bv = bias[n];
            for (int r = 0; r < 8; ++r) {
                int m = m0 + wm * 32 + mt * 16 + r + hi * 8;  // C layout rows
                float val = acc[mt][nt][r] + bv;
                size_t idx = (size_t)m * N + n;
                if (mode == 0) {
                    outBf[idx] = f2bf(val);
                } else if (mode == 1) {
                    outF[idx] = val + res[idx];
                } else {
                    float ge = 0.5f * val * (1.f + erff(val * 0.70710678118f));
                    outBf[idx] = f2bf(ge);
                }
            }
        }
}

// ---------------------------------------------------------------------------
// Flash attention: wave = one 16-row q tile of one (b,h). WG = 8 waves share
// double-buffered 32-key K/V LDS tiles (async-filled). scores K=64 (2 chained
// WMMA), online softmax in 16-lane halves, P relayout via per-wave LDS
// scratch, ctx via 4 WMMAs. q/k/v/ctx bf16 in [B,S,HID], heads interleaved.
// ---------------------------------------------------------------------------
__global__ void __launch_bounds__(256)
attn_kernel(const unsigned short* __restrict__ q,
            const unsigned short* __restrict__ k,
            const unsigned short* __restrict__ v,
            const float* __restrict__ mask,
            unsigned short* __restrict__ ctx) {
    __shared__ unsigned short Kt[2][32][72];  // 32 keys x 64 dh (pad->72)
    __shared__ unsigned short Vt[2][32][72];
    __shared__ unsigned short Ps[8][16 * 40]; // per-wave P scratch (16x32, pad 40)

    int wg = blockIdx.x;
    int b  = wg >> 8;            // 16 heads * 16 qblocks per batch
    int h  = (wg >> 4) & 15;
    int qb = wg & 15;
    int tid = threadIdx.x, w = tid >> 5, lane = tid & 31;
    int l16 = lane & 15, hi = lane >> 4;
    int s0 = qb * 128 + w * 16;

    // Q A-fragments: two 16x32 chunks covering dh = 64
    BFrag qf[2];
    {
        const unsigned short* qbase =
            q + ((size_t)(b * SEQ + s0 + l16)) * HID + h * 64;
        for (int c = 0; c < 2; ++c)
            for (int r = 0; r < 8; ++r) {
                int kk = (r < 4 ? 0 : 16) + (hi ? 8 : 0) + 2 * (r & 3);
                qf[c].u[r] = *(const unsigned*)(qbase + c * 32 + kk);
            }
    }

    v8f o[4] = {v8f{}, v8f{}, v8f{}, v8f{}};
    float mrow[8], lrow[8];
    for (int r = 0; r < 8; ++r) { mrow[r] = -3.0e38f; lrow[r] = 0.f; }

    const size_t kvbase = ((size_t)b * SEQ) * HID + h * 64;
    int ldk = tid >> 3, ldc = (tid & 7) * 8;   // K/V tile cooperative load

    auto issueKV = [&](int j, int buf) {
        cp_async16(k + kvbase + (size_t)(j * 32 + ldk) * HID + ldc, &Kt[buf][ldk][ldc]);
        cp_async16(v + kvbase + (size_t)(j * 32 + ldk) * HID + ldc, &Vt[buf][ldk][ldc]);
    };

    issueKV(0, 0);
    for (int j = 0; j < SEQ / 32; ++j) {
        int buf = j & 1;
        __syncthreads();
        if (j + 1 < SEQ / 32) { issueKV(j + 1, buf ^ 1); ASYNC_WAIT(2); }
        else                  { ASYNC_WAIT(0); }
        __syncthreads();

        // ---- scores: 16 q rows x 32 keys, K = 64 ----
        v8f sf0 = {}, sf1 = {};
        int kb = hi ? 8 : 0;
        for (int c = 0; c < 2; ++c) {
            BFrag kb0, kb1;
            for (int r = 0; r < 8; ++r) {
                kb0.u[r] = *(const unsigned*)&Kt[buf][l16][c * 32 + (r < 4 ? 0 : 16) + kb + 2 * (r & 3)];
                kb1.u[r] = *(const unsigned*)&Kt[buf][16 + l16][c * 32 + (r < 4 ? 0 : 16) + kb + 2 * (r & 3)];
            }
            sf0 = wmma_bf16(qf[c], kb0, sf0);
            sf1 = wmma_bf16(qf[c], kb1, sf1);
        }

        float mv0 = (mask[b * SEQ + j * 32 + l16]      - 1.f) * 10000.f;
        float mv1 = (mask[b * SEQ + j * 32 + 16 + l16] - 1.f) * 10000.f;

        // ---- online softmax (per-row stats live per (lane-half, VGPR r)) ----
        float p0[8], p1[8];
        for (int r = 0; r < 8; ++r) {
            float a  = sf0[r] * 0.125f + mv0;
            float bb = sf1[r] * 0.125f + mv1;
            float mx   = halfmax(fmaxf(a, bb));
            float mnew = fmaxf(mrow[r], mx);
            float corr = __expf(mrow[r] - mnew);
            mrow[r] = mnew;
            p0[r] = __expf(a - mnew);
            p1[r] = __expf(bb - mnew);
            lrow[r] = lrow[r] * corr + halfsum(p0[r] + p1[r]);
            o[0][r] *= corr; o[1][r] *= corr; o[2][r] *= corr; o[3][r] *= corr;
        }

        // ---- relayout P (C-layout) -> A-layout through per-wave LDS ----
        unsigned short* ps = Ps[w];
        for (int r = 0; r < 8; ++r) {
            int row = r + hi * 8;
            ps[row * 40 + l16]      = f2bf(p0[r]);
            ps[row * 40 + 16 + l16] = f2bf(p1[r]);
        }
        asm volatile("s_wait_dscnt 0" ::: "memory");  // wave-internal LDS RAW
        BFrag pf;
        for (int r = 0; r < 8; ++r) {
            int kk = (r < 4 ? 0 : 16) + (hi ? 8 : 0) + 2 * (r & 3);
            pf.u[r] = *(const unsigned*)&ps[l16 * 40 + kk];
        }

        // ---- ctx += P(16x32) * V(32x64) ----
        for (int nt = 0; nt < 4; ++nt) {
            BFrag vb;
            for (int r = 0; r < 8; ++r) {
                unsigned lo = Vt[buf][(r < 4 ? 0 : 16) + kb + 2 * (r & 3)][nt * 16 + l16];
                unsigned hv = Vt[buf][(r < 4 ? 0 : 16) + kb + 2 * (r & 3) + 1][nt * 16 + l16];
                vb.u[r] = lo | (hv << 16);
            }
            o[nt] = wmma_bf16(pf, vb, o[nt]);
        }
    }

    for (int nt = 0; nt < 4; ++nt)
        for (int r = 0; r < 8; ++r) {
            int row = s0 + r + hi * 8;
            int col = h * 64 + nt * 16 + l16;
            float val = o[nt][r] / lrow[r];
            ctx[((size_t)(b * SEQ + row)) * HID + col] = f2bf(val);
        }
}

// ---------------------------------------------------------------------------
// Host-side launch sequence
// ---------------------------------------------------------------------------
extern "C" void kernel_launch(void* const* d_in, const int* in_sizes, int n_in,
                              void* d_out, int out_size, void* d_ws, size_t ws_size,
                              hipStream_t stream) {
    (void)in_sizes; (void)n_in; (void)out_size; (void)ws_size;
    const float* x    = (const float*)d_in[0];
    const float* mask = (const float*)d_in[1];
    const float* Wq = (const float*)d_in[2];  const float* bq = (const float*)d_in[3];
    const float* Wk = (const float*)d_in[4];  const float* bk = (const float*)d_in[5];
    const float* Wv = (const float*)d_in[6];  const float* bv = (const float*)d_in[7];
    const float* Wo = (const float*)d_in[8];  const float* bo = (const float*)d_in[9];
    const float* W1 = (const float*)d_in[10]; const float* b1 = (const float*)d_in[11];
    const float* W2 = (const float*)d_in[12]; const float* b2 = (const float*)d_in[13];
    const float* g1  = (const float*)d_in[14]; const float* be1 = (const float*)d_in[15];
    const float* g2  = (const float*)d_in[16]; const float* be2 = (const float*)d_in[17];

    const size_t MB = 1u << 20;
    char* ws = (char*)d_ws;
    unsigned short* wq_bf = (unsigned short*)(ws + 0 * MB);   // 2 MB each
    unsigned short* wk_bf = (unsigned short*)(ws + 2 * MB);
    unsigned short* wv_bf = (unsigned short*)(ws + 4 * MB);
    unsigned short* wo_bf = (unsigned short*)(ws + 6 * MB);
    unsigned short* w1_bf = (unsigned short*)(ws + 8 * MB);   // 8 MB
    unsigned short* w2_bf = (unsigned short*)(ws + 16 * MB);  // 8 MB
    unsigned short* hn1   = (unsigned short*)(ws + 24 * MB);  // 8 MB
    unsigned short* qb_   = (unsigned short*)(ws + 32 * MB);  // 8 MB
    unsigned short* kb_   = (unsigned short*)(ws + 40 * MB);  // 8 MB
    unsigned short* vb_   = (unsigned short*)(ws + 48 * MB);  // 8 MB
    unsigned short* ctxb  = (unsigned short*)(ws + 56 * MB);  // 8 MB
    float*          x1    = (float*)(ws + 64 * MB);           // 16 MB
    unsigned short* hn2   = (unsigned short*)(ws + 80 * MB);  // 8 MB
    unsigned short* ff1   = (unsigned short*)(ws + 88 * MB);  // 32 MB
    float* out = (float*)d_out;

    const int M = NB * SEQ;        // 4096
    const int H1M = HID * HID;     // 1M elems
    const int FFM = 4 * HID * HID; // 4M elems

    // 1) weights -> bf16
    cvt_bf16_kernel<<<H1M / 256, 256, 0, stream>>>(Wq, wq_bf, H1M);
    cvt_bf16_kernel<<<H1M / 256, 256, 0, stream>>>(Wk, wk_bf, H1M);
    cvt_bf16_kernel<<<H1M / 256, 256, 0, stream>>>(Wv, wv_bf, H1M);
    cvt_bf16_kernel<<<H1M / 256, 256, 0, stream>>>(Wo, wo_bf, H1M);
    cvt_bf16_kernel<<<FFM / 256, 256, 0, stream>>>(W1, w1_bf, FFM);
    cvt_bf16_kernel<<<FFM / 256, 256, 0, stream>>>(W2, w2_bf, FFM);

    // 2) LN1
    ln_kernel<<<M, 256, 0, stream>>>(x, g1, be1, hn1);

    // 3) QKV projections (mode 0: bf16 out)
    dim3 gP(HID / 64, M / 128);
    gemm_kernel<<<gP, 256, 0, stream>>>(hn1, wq_bf, bq, qb_, nullptr, nullptr, M, HID, HID, 0);
    gemm_kernel<<<gP, 256, 0, stream>>>(hn1, wk_bf, bk, kb_, nullptr, nullptr, M, HID, HID, 0);
    gemm_kernel<<<gP, 256, 0, stream>>>(hn1, wv_bf, bv, vb_, nullptr, nullptr, M, HID, HID, 0);

    // 4) attention
    attn_kernel<<<NB * 16 * (SEQ / 128), 256, 0, stream>>>(qb_, kb_, vb_, mask, ctxb);

    // 5) O projection + residual (mode 1: fp32 out)
    gemm_kernel<<<gP, 256, 0, stream>>>(ctxb, wo_bf, bo, nullptr, x1, x, M, HID, HID, 1);

    // 6) LN2
    ln_kernel<<<M, 256, 0, stream>>>(x1, g2, be2, hn2);

    // 7) FF1 + GELU (mode 2)
    dim3 gF1(4 * HID / 64, M / 128);
    gemm_kernel<<<gF1, 256, 0, stream>>>(hn2, w1_bf, b1, ff1, nullptr, nullptr, M, 4 * HID, HID, 2);

    // 8) FF2 + residual -> d_out (mode 1)
    gemm_kernel<<<gP, 256, 0, stream>>>(ff1, w2_bf, b2, nullptr, out, x1, M, HID, 4 * HID, 1);
}
